// MORP_72026601554059
// MI455X (gfx1250) — matrix operand order repaired and checked by reading, not compiled, not run on hardware
//
#include <hip/hip_runtime.h>
#include <hip/hip_bf16.h>

typedef __attribute__((ext_vector_type(16))) _Float16 v16h;
typedef __attribute__((ext_vector_type(8)))  float    v8f;

#define N_FFT 4096
#define NROW  8192
#define M_SP  4097
#define PI_F  3.14159265358979323846f

__device__ __forceinline__ float eluf(float x) {
    return x > 0.0f ? x : (__expf(x) - 1.0f);
}

// ---------------------------------------------------------------------------
// Kernel 1: g MLP on k values: 1 -> 64 -> 64 -> 2  (ELU on hidden layers)
// Tiny (17 MFLOP); plain VALU. Writes gk_re / gk_im into workspace.
// ---------------------------------------------------------------------------
__global__ void __launch_bounds__(256) morp_g_mlp(
    const float* __restrict__ kv,
    const float* __restrict__ w1, const float* __restrict__ b1,
    const float* __restrict__ w2, const float* __restrict__ b2,
    const float* __restrict__ w3, const float* __restrict__ b3,
    float* __restrict__ gre, float* __restrict__ gim, int M) {
    int t = blockIdx.x * blockDim.x + threadIdx.x;
    if (t >= M) return;
    float x = kv[t];
    float h1[64];
#pragma unroll
    for (int j = 0; j < 64; ++j) h1[j] = eluf(fmaf(x, w1[j], b1[j]));
    float re = b3[0], im = b3[1];
    for (int n = 0; n < 64; ++n) {
        float s = b2[n];
#pragma unroll
        for (int j = 0; j < 64; ++j) s = fmaf(h1[j], w2[j * 64 + n], s);
        s = eluf(s);
        re = fmaf(s, w3[n * 2 + 0], re);
        im = fmaf(s, w3[n * 2 + 1], im);
    }
    gre[t] = re;
    gim[t] = im;
}

// ---------------------------------------------------------------------------
// Stockham radix-2 stage loop (interleaved complex in LDS, ping-pong).
// After the loop, *ps points at the result buffer.
// ---------------------------------------------------------------------------
__device__ __forceinline__ void fft_stages(float** ps, float** pd, float sgn, int tid) {
    float* s = *ps;
    float* d = *pd;
    for (int width = 1; width < N_FFT; width <<= 1) {
        __syncthreads();
        float wstep = sgn * PI_F / (float)width;
        for (int t = tid; t < N_FFT / 2; t += 512) {
            int j = t & (width - 1);
            float2 a = ((const float2*)s)[t];
            float2 b = ((const float2*)s)[t + N_FFT / 2];
            float sn, cs;
            __sincosf(wstep * (float)j, &sn, &cs);
            float wbr = cs * b.x - sn * b.y;
            float wbi = cs * b.y + sn * b.x;
            int o = 2 * t - j;
            ((float2*)d)[o]         = make_float2(a.x + wbr, a.y + wbi);
            ((float2*)d)[o + width] = make_float2(a.x - wbr, a.y - wbi);
        }
        float* tmp = s; s = d; d = tmp;
    }
    *ps = s;
    *pd = d;
}

// ---------------------------------------------------------------------------
// Kernel 2: one workgroup per row (2048 rows, 512 threads = 16 waves).
//   Phase 1: pointwise h-MLP via v_wmma_f32_16x16x32_f16 -> hu into LDS
//   Phase 2: forward complex FFT-4096 (packed real-8192)
//   Phase 3: rfft unpack * g(k), irfft repack (conjugate-pair fused pass)
//   Phase 4: inverse FFT, scale 1/4096, store row
// ---------------------------------------------------------------------------
__global__ void __launch_bounds__(512) morp_row_kernel(
    const float* __restrict__ u,
    const float* __restrict__ hw1, const float* __restrict__ hb1,
    const float* __restrict__ hw2, const float* __restrict__ hb2,
    const float* __restrict__ hw3, const float* __restrict__ hb3p,
    const float* __restrict__ gre, const float* __restrict__ gim,
    float* __restrict__ out) {
    __shared__ __align__(16) float bufA[2 * N_FFT];
    __shared__ __align__(16) float bufB[2 * N_FFT];

    const int tid  = threadIdx.x;
    const int lane = tid & 31;
    const int wv   = tid >> 5;
    const int row  = blockIdx.x;
    const float* urow = u   + (size_t)row * NROW;
    float*       orow = out + (size_t)row * NROW;
    const float hb3 = hb3p[0];

    // ---- W2 as 8 constant f16 B-fragments: B[k_chunk][n_tile], K=64, N=64.
// B 32x16 layout: lane holds column N=lane%16; element j holds
// K = 8*(lane/16) + 16*(j/8) + (j%8) (same order as A).
    v16h Bf[2][4];
#pragma unroll
    for (int c = 0; c < 2; ++c) {
        int krow = c * 32 + ((lane >> 4) << 3);
#pragma unroll
        for (int t4 = 0; t4 < 4; ++t4)
#pragma unroll
            for (int j = 0; j < 16; ++j)
                Bf[c][t4][j] = (_Float16)hw2[(krow + ((j >> 3) << 4) + (j & 7)) * 64 + t4 * 16 + (lane & 15)];
    }
    // per-lane layer-1 weights matching the 16-bit A 16x32 layout:
    // k(lane,j) = 8*(lane/16) + 16*(j/8) + (j%8)   (and +32 for chunk 1)
    float w1A[16], b1A[16], w1B[16], b1B[16];
#pragma unroll
    for (int j = 0; j < 16; ++j) {
        int k = ((lane >> 4) << 3) + ((j >> 3) << 4) + (j & 7);
        w1A[j] = hw1[k];      b1A[j] = hb1[k];
        w1B[j] = hw1[k + 32]; b1B[j] = hb1[k + 32];
    }
    float w3v[4], b2v[4];
#pragma unroll
    for (int t4 = 0; t4 < 4; ++t4) {
        int n = t4 * 16 + (lane & 15);
        w3v[t4] = hw3[n];
        b2v[t4] = hb2[n];
    }

    // ================= Phase 1: h-MLP, 16 elements/wave/iteration ==========
    const int ebase = wv * 512;
    for (int it = 0; it < 32; ++it) {
        const int e0 = ebase + it * 16;
        __builtin_prefetch(urow + e0 + 16, 0, 3);
        float x = urow[e0 + (lane & 15)];
        v16h a0, a1;
#pragma unroll
        for (int j = 0; j < 16; ++j) {
            a0[j] = (_Float16)eluf(fmaf(x, w1A[j], b1A[j]));
            a1[j] = (_Float16)eluf(fmaf(x, w1B[j], b1B[j]));
        }
        v8f acc[4];
#pragma unroll
        for (int t4 = 0; t4 < 4; ++t4) {
            v8f z = {};
            z = __builtin_amdgcn_wmma_f32_16x16x32_f16(false, a0, false, Bf[0][t4],
                                                       (short)0, z, false, false);
            z = __builtin_amdgcn_wmma_f32_16x16x32_f16(false, a1, false, Bf[1][t4],
                                                       (short)0, z, false, false);
            acc[t4] = z;
        }
        // epilogue: z2 + b2 -> ELU -> dot w3 ; row m = v + 8*(lane/16),
        // columns n = 16*t4 + (lane&15) -> reduce over the 16-lane half.
        float sv[8];
#pragma unroll
        for (int v = 0; v < 8; ++v) {
            float p = 0.0f;
#pragma unroll
            for (int t4 = 0; t4 < 4; ++t4)
                p = fmaf(eluf(acc[t4][v] + b2v[t4]), w3v[t4], p);
            p += __shfl_xor(p, 1, 32);
            p += __shfl_xor(p, 2, 32);
            p += __shfl_xor(p, 4, 32);
            p += __shfl_xor(p, 8, 32);
            sv[v] = p + hb3;
        }
        if ((lane & 15) < 8) {
            int sel = lane & 7;
            float r = sv[0];
#pragma unroll
            for (int q = 1; q < 8; ++q) r = (sel == q) ? sv[q] : r;
            bufA[e0 + ((lane >> 4) << 3) + sel] = r;   // hu, natural order
        }
    }

    // ================= Phase 2: forward FFT (packed real -> Z) =============
    float* s = bufA;
    float* d = bufB;
    fft_stages(&s, &d, -1.0f, tid);   // result Z in *s
    __syncthreads();

    // ================= Phase 3: X = rfft(hu); Y = g*X; W = irfft-pack(Y) ===
    for (int k = tid; k <= N_FFT / 2; k += 512) {
        int nkm = (N_FFT - k) & (N_FFT - 1);
        float2 Zk  = ((const float2*)s)[k];
        float2 Znk = ((const float2*)s)[nkm];
        // Xe = (Z[k]+conj(Z[N-k]))/2 ; Xo = -i/2 (Z[k]-conj(Z[N-k]))
        float xer  = 0.5f * (Zk.x + Znk.x);
        float xei  = 0.5f * (Zk.y - Znk.y);
        float xor_ = 0.5f * (Zk.y + Znk.y);
        float xoi  = -0.5f * (Zk.x - Znk.x);
        float sn, cs;
        __sincosf(-PI_F * (float)k / (float)N_FFT, &sn, &cs);  // t = e^{-i pi k/N}
        float txr = cs * xor_ - sn * xoi;
        float txi = cs * xoi + sn * xor_;
        float Xkr = xer + txr, Xki = xei + txi;        // X[k]
        float Xnr = xer - txr, Xni = -(xei - txi);     // X[N-k] = conj(Xe - t*Xo)
        // multiply by spectral operator g
        float gkr = gre[k],        gki = gim[k];
        float gnr = gre[N_FFT - k], gni = gim[N_FFT - k];
        float Ykr = gkr * Xkr - gki * Xki, Yki = gkr * Xki + gki * Xkr;
        float Ynr = gnr * Xnr - gni * Xni, Yni = gnr * Xni + gni * Xnr;
        // W[k] = Ye + i*Yo ; Ye=(Yk+conj(Yn))/2, Yo=e^{+i pi k/N}(Yk-conj(Yn))/2
        float yer = 0.5f * (Ykr + Ynr), yei = 0.5f * (Yki - Yni);
        float dr  = 0.5f * (Ykr - Ynr), di  = 0.5f * (Yki + Yni);
        float uor = cs * dr + sn * di;     // e^{+i pi k/N} = (cs, -sn)
        float uoi = cs * di - sn * dr;
        ((float2*)d)[k] = make_float2(yer - uoi, yei + uor);
        if (k > 0 && k < N_FFT / 2) {
            // W[N-k] simplifies to (yer + uoi) + i*(uor - yei)
            ((float2*)d)[N_FFT - k] = make_float2(yer + uoi, uor - yei);
        }
    }

    // ================= Phase 4: inverse FFT of W, scale, store =============
    { float* tmp = s; s = d; d = tmp; }   // s = W buffer
    fft_stages(&s, &d, 1.0f, tid);        // result in *s (un-normalized)
    __syncthreads();
    const float inv_n = 1.0f / (float)N_FFT;
    for (int i = tid; i < NROW; i += 512)
        orow[i] = inv_n * s[i];           // interleaved z == real samples
}

// ---------------------------------------------------------------------------
extern "C" void kernel_launch(void* const* d_in, const int* in_sizes, int n_in,
                              void* d_out, int out_size, void* d_ws, size_t ws_size,
                              hipStream_t stream) {
    const float* u   = (const float*)d_in[0];
    const float* kv  = (const float*)d_in[1];
    const float* hw1 = (const float*)d_in[2];
    const float* hb1 = (const float*)d_in[3];
    const float* hw2 = (const float*)d_in[4];
    const float* hb2 = (const float*)d_in[5];
    const float* hw3 = (const float*)d_in[6];
    const float* hb3 = (const float*)d_in[7];
    const float* gw1 = (const float*)d_in[8];
    const float* gb1 = (const float*)d_in[9];
    const float* gw2 = (const float*)d_in[10];
    const float* gb2 = (const float*)d_in[11];
    const float* gw3 = (const float*)d_in[12];
    const float* gb3 = (const float*)d_in[13];

    float* gre = (float*)d_ws;          // M_SP floats
    float* gim = gre + 4160;            // padded offset

    morp_g_mlp<<<(M_SP + 255) / 256, 256, 0, stream>>>(
        kv, gw1, gb1, gw2, gb2, gw3, gb3, gre, gim, M_SP);

    morp_row_kernel<<<2048, 512, 0, stream>>>(
        u, hw1, hb1, hw2, hb2, hw3, hb3, gre, gim, (float*)d_out);
}
